// update_e_55387898250017
// MI455X (gfx1250) — compile-verified, matrix operand order checked by reading, not run on hardware
//
#include <hip/hip_runtime.h>
#include <hip/hip_bf16.h>

// ---------------------------------------------------------------------------
// Problem constants (match reference)
// ---------------------------------------------------------------------------
#define E_N 262144
#define T_N 1048576
#define H_N 128
#define IE_N 64
#define NR_N 6
#define NSR_N 42   // NS*NR
#define BD_N 8
#define BA_N 8

typedef __attribute__((ext_vector_type(16))) __bf16 v16bf;
typedef __attribute__((ext_vector_type(8)))  float  v8f;
typedef __attribute__((ext_vector_type(4)))  unsigned int u32x4;
typedef __attribute__((ext_vector_type(8)))  int          i32x8;
typedef __attribute__((ext_vector_type(4)))  int          i32x4;

#if defined(__has_builtin)
#  if __has_builtin(__builtin_amdgcn_tensor_load_to_lds)
#    define HAVE_TDM 1
#  else
#    define HAVE_TDM 0
#  endif
#else
#  define HAVE_TDM 0
#endif

// ---------------------------------------------------------------------------
// Weight convert+transpose: W [K][N] f32 -> Wt [N][K] bf16
// ---------------------------------------------------------------------------
__global__ void conv_wt_kernel(const float* __restrict__ W, __bf16* __restrict__ Wt,
                               int K, int N) {
    int i = blockIdx.x * 256 + threadIdx.x;
    if (i < K * N) {
        int k = i / N, n = i % N;
        Wt[n * K + k] = (__bf16)W[i];
    }
}

// ---------------------------------------------------------------------------
// segment_sum(x1, idx_ji) -> acc [E,H]  (float4 per thread, fp32 atomics)
// ---------------------------------------------------------------------------
__global__ __launch_bounds__(256) void seg_sum_kernel(const float* __restrict__ x1,
                                                      const int* __restrict__ idx_ji,
                                                      float* __restrict__ acc) {
    size_t i = (size_t)blockIdx.x * 256 + threadIdx.x;  // over T * (H/4)
    int t  = (int)(i >> 5);
    int c4 = ((int)i & 31) * 4;
    const float4 v = *(const float4*)(x1 + (size_t)t * H_N + c4);
    float* dst = acc + (size_t)idx_ji[t] * H_N + c4;
    atomicAdd(dst + 0, v.x);
    atomicAdd(dst + 1, v.y);
    atomicAdd(dst + 2, v.z);
    atomicAdd(dst + 3, v.w);
}

// ---------------------------------------------------------------------------
// Fused triplet stage:
//   sbf_p = (sbf @ W_sbf1) @ W_sbf2          [T,64]  (never materialized)
//   v     = x_kj_down[idx_kj] * sbf_p
//   agg[idx_ji] += v                          (fp32 atomics, L2-resident target)
// block = 256 threads = 4 triplets x 64 channels
// ---------------------------------------------------------------------------
__global__ __launch_bounds__(256) void triplet_kernel(
    const float* __restrict__ sbf,    // [T,42]
    const float* __restrict__ W_sbf1, // [42,8]
    const float* __restrict__ W_sbf2, // [8,64]
    const float* __restrict__ xkj,    // [E,64]
    const int*   __restrict__ idx_kj,
    const int*   __restrict__ idx_ji,
    float*       __restrict__ agg)    // [E,64]
{
    __shared__ float sW1[NSR_N * BA_N];
    __shared__ float sW2[BA_N * IE_N];
    __shared__ float sT8[4][BA_N];
    __shared__ float sSbf[4][NSR_N];

    int tid = threadIdx.x;
    for (int i = tid; i < NSR_N * BA_N; i += 256) sW1[i] = W_sbf1[i];
    for (int i = tid; i < BA_N * IE_N; i += 256) sW2[i] = W_sbf2[i];
    __syncthreads();

    int y = tid >> 6;   // triplet within block (0..3)
    int x = tid & 63;   // channel (0..63)
    size_t t = (size_t)blockIdx.x * 4 + y;

    if (x < NSR_N) sSbf[y][x] = sbf[t * NSR_N + x];
    __syncthreads();
    if (x < BA_N) {
        float s = 0.f;
        for (int p = 0; p < NSR_N; ++p) s += sSbf[y][p] * sW1[p * BA_N + x];
        sT8[y][x] = s;
    }
    __syncthreads();

    float v = 0.f;
    for (int q = 0; q < BA_N; ++q) v += sT8[y][q] * sW2[q * IE_N + x];

    int ekj = idx_kj[t];
    int eji = idx_ji[t];
    v *= xkj[(size_t)ekj * IE_N + x];
    atomicAdd(&agg[(size_t)eji * IE_N + x], v);
}

// ---------------------------------------------------------------------------
// WMMA GEMM with fused epilogue:
//   Out = epilogue( silu( X[M,K] @ W[K,N] + bias[N] ) )
//   MODE 0: plain
//   MODE 1: * rbf(row,col)   where rbf = (rbf0 @ W_rbf1) @ W_rbf2 computed on the fly
//   MODE 2: + extra[M,N]
// Block = 256 threads = 8 waves; wave -> 16 rows x N cols; K-step = 32.
// Weights pulled into LDS as bf16 transposed [N][K]:
//   - preferred: one TENSOR_LOAD_TO_LDS DMA issued by wave 0 (TENSORcnt path)
//   - fallback:  cooperative uint4 loads
// ---------------------------------------------------------------------------
template<int K, int N, int MODE>
__global__ __launch_bounds__(256) void gemm_act_kernel(
    const float*  __restrict__ X,       // [M,K]
    const __bf16* __restrict__ Wt,      // [N][K] bf16 (pre-transposed)
    const float*  __restrict__ bias,    // [N]
    const float*  __restrict__ extra,   // [M,N] (MODE 2)
    const float*  __restrict__ rbf0,    // [M,6] (MODE 1)
    const float*  __restrict__ W_rbf1,  // [6,8] (MODE 1)
    const float*  __restrict__ W_rbf2,  // [8,N] (MODE 1)
    float*        __restrict__ Out,     // [M,N]
    int M)
{
    constexpr int NT = N / 16;           // number of 16-col tiles
    __shared__ __align__(16) __bf16 Blds[N * K];

    const int tid = threadIdx.x;
#if HAVE_TDM
    // Tensor Data Mover: single DMA of the N*K bf16 weight tile into LDS.
    // D# descriptor per CDNA5 ISA (1-D tile, 4B elems).
    if (tid < 32) {
        constexpr unsigned NELEM = (unsigned)(N * K / 2);   // 4-byte elements
        unsigned long long ga = (unsigned long long)(const void*)Wt;
        unsigned lds_off = (unsigned)(uintptr_t)(&Blds[0]);
        u32x4 g0;
        g0[0] = 1u;                                          // count=1 (valid D#)
        g0[1] = lds_off;                                     // lds_addr
        g0[2] = (unsigned)(ga & 0xFFFFFFFFu);                // global_addr[31:0]
        g0[3] = (unsigned)((ga >> 32) & 0x1FFFFFFu)          // global_addr[56:32]
                | (2u << 30);                                // type=2 ("image")
        i32x8 g1 = {};
        g1[0] = (int)(2u << 16);                             // data_size=2 (4B)
        g1[1] = (int)((NELEM & 0xFFFFu) << 16);              // tensor_dim0[15:0]
        g1[2] = (int)(NELEM >> 16);                          // tensor_dim0[31:16]
        g1[3] = (int)(NELEM << 16);                          // tile_dim0
        g1[5] = (int)NELEM;                                  // tensor_dim0_stride[31:0]
        i32x4 gz = {};
#if __clang_major__ >= 23
        i32x8 gz8 = {};
        __builtin_amdgcn_tensor_load_to_lds(g0, g1, gz, gz, gz8, 0);
#else
        __builtin_amdgcn_tensor_load_to_lds(g0, g1, gz, gz, 0);
#endif
        __builtin_amdgcn_s_wait_tensorcnt(0);
    }
    __syncthreads();
#else
    {   // cooperative LDS fill of weights (uint4 = 8 bf16 per load)
        const uint4* src = (const uint4*)Wt;
        uint4* dst = (uint4*)Blds;
        for (int i = tid; i < N * K / 8; i += 256) dst[i] = src[i];
    }
    __syncthreads();
#endif

    const int wave = tid >> 5;
    const int lane = tid & 31;
    const int half = lane >> 4;          // 0: lanes 0-15, 1: lanes 16-31
    const int l16  = lane & 15;
    const int rowBase = blockIdx.x * 128 + wave * 16;
    if (rowBase >= M) return;

    v8f acc[NT] = {};

    const int arow = rowBase + l16;
    __builtin_prefetch(X + (size_t)arow * K, 0, 0);   // global_prefetch_b8

    for (int k0 = 0; k0 < K; k0 += 32) {
        // --- A fragment: 16-bit A 16x32 layout ---
        // lane<16 : K = {kb..kb+7, kb+16..kb+23} with kb=k0
        // lane>=16: kb = k0+8  -> K = {8..15, 24..31}
        const int kb = k0 + 8 * half;
        const float4* xp = (const float4*)(X + (size_t)arow * K + kb);
        float4 a0 = xp[0];
        float4 a1 = xp[1];
        float4 a2 = xp[4];   // +16 floats
        float4 a3 = xp[5];
        v16bf a;
        a[0]=(__bf16)a0.x; a[1]=(__bf16)a0.y; a[2]=(__bf16)a0.z; a[3]=(__bf16)a0.w;
        a[4]=(__bf16)a1.x; a[5]=(__bf16)a1.y; a[6]=(__bf16)a1.z; a[7]=(__bf16)a1.w;
        a[8]=(__bf16)a2.x; a[9]=(__bf16)a2.y; a[10]=(__bf16)a2.z; a[11]=(__bf16)a2.w;
        a[12]=(__bf16)a3.x; a[13]=(__bf16)a3.y; a[14]=(__bf16)a3.z; a[15]=(__bf16)a3.w;

        // --- B fragments from LDS (transposed weights: column-contiguous K) ---
        #pragma unroll
        for (int j = 0; j < NT; ++j) {
            const int bcol = j * 16 + l16;
            const uint4* bp = (const uint4*)(Blds + (size_t)bcol * K + kb);
            uint4 b0 = bp[0];   // K kb..kb+7
            uint4 b1 = bp[2];   // K kb+16..kb+23
            v16bf b;
            reinterpret_cast<uint4*>(&b)[0] = b0;
            reinterpret_cast<uint4*>(&b)[1] = b1;
            acc[j] = __builtin_amdgcn_wmma_f32_16x16x32_bf16(
                false, a, false, b, (short)0, acc[j], false, false);
        }
    }

    // --- epilogue: C/D layout -> col = l16+16j, rows = rowBase + 8*half + r ---
    const int outRow0 = rowBase + half * 8;
    for (int r = 0; r < 8; ++r) {
        const int row = outRow0 + r;
        float t8[8];
        if (MODE == 1) {
            const float* rp = rbf0 + (size_t)row * NR_N;
            float r6[NR_N];
            #pragma unroll
            for (int p = 0; p < NR_N; ++p) r6[p] = rp[p];
            #pragma unroll
            for (int q = 0; q < BD_N; ++q) {
                float s = 0.f;
                #pragma unroll
                for (int p = 0; p < NR_N; ++p) s += r6[p] * W_rbf1[p * BD_N + q];
                t8[q] = s;
            }
        }
        #pragma unroll
        for (int j = 0; j < NT; ++j) {
            const int col = j * 16 + l16;
            float v = acc[j][r] + bias[col];
            v = v / (1.f + __expf(-v));   // silu
            if (MODE == 1) {
                float rb = 0.f;
                #pragma unroll
                for (int q = 0; q < BD_N; ++q) rb += t8[q] * W_rbf2[q * N + col];
                v *= rb;
            }
            if (MODE == 2) {
                v += extra[(size_t)row * N + col];
            }
            Out[(size_t)row * N + col] = v;
        }
    }
}

// ---------------------------------------------------------------------------
// e2 = (rbf0 @ W_rbf) * e1   (on-the-fly, W_rbf in LDS)
// ---------------------------------------------------------------------------
__global__ __launch_bounds__(256) void e2_kernel(const float* __restrict__ rbf0,
                                                 const float* __restrict__ W_rbf,
                                                 const float* __restrict__ e1,
                                                 float* __restrict__ e2) {
    __shared__ float sW[NR_N * H_N];
    int tid = threadIdx.x;
    for (int i = tid; i < NR_N * H_N; i += 256) sW[i] = W_rbf[i];
    __syncthreads();
    size_t i = (size_t)blockIdx.x * 256 + tid;
    int row = (int)(i >> 7);
    int col = (int)i & 127;
    const float* rp = rbf0 + (size_t)row * NR_N;
    float s = 0.f;
    #pragma unroll
    for (int p = 0; p < NR_N; ++p) s += rp[p] * sW[p * H_N + col];
    e2[i] = s * e1[i];
}

// ---------------------------------------------------------------------------
// Host orchestration
// ---------------------------------------------------------------------------
extern "C" void kernel_launch(void* const* d_in, const int* in_sizes, int n_in,
                              void* d_out, int out_size, void* d_ws, size_t ws_size,
                              hipStream_t stream) {
    const float* x1        = (const float*)d_in[0];
    const float* x_old     = (const float*)d_in[1];
    const float* rbf0      = (const float*)d_in[2];
    const float* sbf       = (const float*)d_in[3];
    const int*   idx_kj    = (const int*)  d_in[4];
    const int*   idx_ji    = (const int*)  d_in[5];
    const float* W_rbf1    = (const float*)d_in[6];
    const float* W_rbf2    = (const float*)d_in[7];
    const float* W_sbf1    = (const float*)d_in[8];
    const float* W_sbf2    = (const float*)d_in[9];
    const float* W_rbf     = (const float*)d_in[10];
    const float* W_kj      = (const float*)d_in[11];
    const float* b_kj      = (const float*)d_in[12];
    const float* W_ji      = (const float*)d_in[13];
    const float* b_ji      = (const float*)d_in[14];
    const float* W_connect = (const float*)d_in[15];
    const float* b_connect = (const float*)d_in[16];
    const float* W_get_up  = (const float*)d_in[17];
    const float* b_get_up  = (const float*)d_in[18];
    const float* W_down    = (const float*)d_in[19];
    const float* b_down    = (const float*)d_in[20];
    const float* W_up      = (const float*)d_in[21];
    const float* b_up      = (const float*)d_in[22];
    const float* rbW       = (const float*)d_in[23];  // [1,2,128,128]
    const float* rbB       = (const float*)d_in[24];  // [1,2,128]
    const float* W_lin     = (const float*)d_in[25];
    const float* b_lin     = (const float*)d_in[26];
    const float* raW       = (const float*)d_in[27];  // [2,2,128,128]
    const float* raB       = (const float*)d_in[28];  // [2,2,128]

    float* e1_out = (float*)d_out;
    float* e2_out = (float*)d_out + (size_t)E_N * H_N;

    // ---- workspace layout ----
    const size_t EH  = (size_t)E_N * H_N;       // elements
    const size_t E64 = (size_t)E_N * IE_N;
    char* ws = (char*)d_ws;
    float* A    = (float*)ws;                 ws += EH  * sizeof(float);
    float* B    = (float*)ws;                 ws += EH  * sizeof(float);
    float* C    = (float*)ws;                 ws += EH  * sizeof(float);
    float* D64a = (float*)ws;                 ws += E64 * sizeof(float);
    float* D64b = (float*)ws;                 ws += E64 * sizeof(float);

    auto take_wt = [&](int K, int N) -> __bf16* {
        __bf16* p = (__bf16*)ws;
        ws += (size_t)K * N * sizeof(__bf16);
        return p;
    };
    __bf16* wt_get_up  = take_wt(128, 128);
    __bf16* wt_ji      = take_wt(128, 128);
    __bf16* wt_kj      = take_wt(128, 128);
    __bf16* wt_down    = take_wt(128, 64);
    __bf16* wt_up      = take_wt(64, 128);
    __bf16* wt_connect = take_wt(128, 128);
    __bf16* wt_rb0     = take_wt(128, 128);
    __bf16* wt_rb1     = take_wt(128, 128);
    __bf16* wt_lin     = take_wt(128, 128);
    __bf16* wt_ra00    = take_wt(128, 128);
    __bf16* wt_ra01    = take_wt(128, 128);
    __bf16* wt_ra10    = take_wt(128, 128);
    __bf16* wt_ra11    = take_wt(128, 128);

    auto conv = [&](const float* W, __bf16* Wt, int K, int N) {
        int n = K * N;
        conv_wt_kernel<<<(n + 255) / 256, 256, 0, stream>>>(W, Wt, K, N);
    };
    conv(W_get_up,  wt_get_up,  128, 128);
    conv(W_ji,      wt_ji,      128, 128);
    conv(W_kj,      wt_kj,      128, 128);
    conv(W_down,    wt_down,    128, 64);
    conv(W_up,      wt_up,      64, 128);
    conv(W_connect, wt_connect, 128, 128);
    conv(rbW + 0 * 16384, wt_rb0, 128, 128);
    conv(rbW + 1 * 16384, wt_rb1, 128, 128);
    conv(W_lin,     wt_lin,     128, 128);
    conv(raW + 0 * 16384, wt_ra00, 128, 128);
    conv(raW + 1 * 16384, wt_ra01, 128, 128);
    conv(raW + 2 * 16384, wt_ra10, 128, 128);
    conv(raW + 3 * 16384, wt_ra11, 128, 128);

    const int GB = E_N / 128;  // gemm grid (rows / 128)

    // 1. A = segment_sum(x1, idx_ji)
    hipMemsetAsync(A, 0, EH * sizeof(float), stream);
    seg_sum_kernel<<<(int)((size_t)T_N * 32 / 256), 256, 0, stream>>>(x1, idx_ji, A);

    // 2. B = silu(A @ W_get_up + b)            [x_up]
    gemm_act_kernel<128,128,0><<<GB,256,0,stream>>>(A, wt_get_up, b_get_up,
        nullptr, nullptr, nullptr, nullptr, B, E_N);
    // 3. C = silu(x_old @ W_ji + b)            [x_ji]
    gemm_act_kernel<128,128,0><<<GB,256,0,stream>>>(x_old, wt_ji, b_ji,
        nullptr, nullptr, nullptr, nullptr, C, E_N);
    // 4. A = silu(x_old @ W_kj + b) * rbf      [x_kj, rbf fused]
    gemm_act_kernel<128,128,1><<<GB,256,0,stream>>>(x_old, wt_kj, b_kj,
        nullptr, rbf0, W_rbf1, W_rbf2, A, E_N);
    // 5. D64a = silu(A @ W_down + b)           [E,64]
    gemm_act_kernel<128,64,0><<<GB,256,0,stream>>>(A, wt_down, b_down,
        nullptr, nullptr, nullptr, nullptr, D64a, E_N);
    // 6. triplet gather/scatter -> D64b
    hipMemsetAsync(D64b, 0, E64 * sizeof(float), stream);
    triplet_kernel<<<T_N / 4, 256, 0, stream>>>(sbf, W_sbf1, W_sbf2, D64a,
                                                idx_kj, idx_ji, D64b);
    // 7. A = silu(D64b @ W_up + b) + C         [e1 = x_ji + x_kj]
    gemm_act_kernel<64,128,2><<<GB,256,0,stream>>>(D64b, wt_up, b_up,
        C, nullptr, nullptr, nullptr, A, E_N);
    // 8. C = silu(A @ W_connect + b) + B       [+ x_up]
    gemm_act_kernel<128,128,2><<<GB,256,0,stream>>>(A, wt_connect, b_connect,
        B, nullptr, nullptr, nullptr, C, E_N);
    // 9. res_before: A = silu(C@W0+b0); B = C + silu(A@W1+b1)
    gemm_act_kernel<128,128,0><<<GB,256,0,stream>>>(C, wt_rb0, rbB + 0,
        nullptr, nullptr, nullptr, nullptr, A, E_N);
    gemm_act_kernel<128,128,2><<<GB,256,0,stream>>>(A, wt_rb1, rbB + 128,
        C, nullptr, nullptr, nullptr, B, E_N);
    // 10. A = silu(B @ W_lin + b) + x_old
    gemm_act_kernel<128,128,2><<<GB,256,0,stream>>>(B, wt_lin, b_lin,
        x_old, nullptr, nullptr, nullptr, A, E_N);
    // 11. res_after[0]: B = silu(A@W+b); C = A + silu(B@W+b)
    gemm_act_kernel<128,128,0><<<GB,256,0,stream>>>(A, wt_ra00, raB + 0,
        nullptr, nullptr, nullptr, nullptr, B, E_N);
    gemm_act_kernel<128,128,2><<<GB,256,0,stream>>>(B, wt_ra01, raB + 128,
        A, nullptr, nullptr, nullptr, C, E_N);
    // 12. res_after[1]: A = silu(C@W+b); e1_out = C + silu(A@W+b)
    gemm_act_kernel<128,128,0><<<GB,256,0,stream>>>(C, wt_ra10, raB + 256,
        nullptr, nullptr, nullptr, nullptr, A, E_N);
    gemm_act_kernel<128,128,2><<<GB,256,0,stream>>>(A, wt_ra11, raB + 384,
        C, nullptr, nullptr, nullptr, e1_out, E_N);
    // 13. e2 = (rbf0 @ W_rbf) * e1
    e2_kernel<<<(int)(EH / 256), 256, 0, stream>>>(rbf0, W_rbf, e1_out, e2_out);
}
